// ConvolutionEmbeddingLayer_29669634081353
// MI455X (gfx1250) — matrix operand order repaired and checked by reading, not compiled, not run on hardware
//
#include <hip/hip_runtime.h>
#include <hip/hip_bf16.h>

typedef __attribute__((ext_vector_type(16))) __bf16 v16bf;
typedef __attribute__((ext_vector_type(8)))  __bf16 v8bf;
typedef __attribute__((ext_vector_type(8)))  float  v8f;
typedef int v4i __attribute__((vector_size(16)));   // matches builtin param type

#define VOCAB 100000
#define SEQ   8192
#define DIM   128
#define KW    5
#define KTOT  (DIM * KW)           // 640
#define NCHUNK (KTOT / 32)         // 20 K-chunks of 32
#define MTILE 128                  // tokens per block
#define HALO_ROWS (MTILE + KW - 1) // 132
#define SA    136                  // LDS A row stride in bf16 (272B -> 4-bank shift/row)
#define SBROW 40                   // LDS B row stride in bf16 (80B  -> 20-dword shift/row)

// gfx1250 async global->LDS path (ASYNCcnt-tracked DMA, no VGPR round trip).
#if __has_builtin(__builtin_amdgcn_global_load_async_to_lds_b128)
#define USE_ASYNC_LDS 1
#else
#define USE_ASYNC_LDS 0
#endif

__device__ __forceinline__ void wait_async0() {
#if __has_builtin(__builtin_amdgcn_s_wait_asynccnt)
    __builtin_amdgcn_s_wait_asynccnt(0);
#else
    asm volatile("s_wait_asynccnt 0x0" ::: "memory");
#endif
}

// ---------------------------------------------------------------------------
// Repack W_conv (out=128, in=128, k=5) f32 -> Bt[n][k] bf16, k = tap*128 + ch.
// Row-contiguous in k so WMMA B fragments are 16B-contiguous loads.
// ---------------------------------------------------------------------------
__global__ void prep_b_kernel(const float* __restrict__ Wc, __bf16* __restrict__ Bt) {
    int idx = blockIdx.x * blockDim.x + threadIdx.x;
    if (idx >= DIM * KTOT) return;
    int n = idx / KTOT;
    int k = idx % KTOT;
    int j = k >> 7;        // tap 0..4
    int c = k & 127;       // input channel
    Bt[idx] = (__bf16)Wc[(n * DIM + c) * KW + j];
}

// Stage one 128x32 bf16 K-chunk of Bt into an LDS buffer (8 KB).
// 512 x 16B pieces across 256 threads.
__device__ __forceinline__ void stage_b_chunk(const __bf16* __restrict__ Bt,
                                              __bf16* __restrict__ dst,
                                              int ki, int tid) {
    #pragma unroll
    for (int rep = 0; rep < 2; ++rep) {
        int cid  = tid + rep * 256;
        int n    = cid >> 2;
        int part = cid & 3;
        const __bf16* g = &Bt[n * KTOT + ki * 32 + part * 8];
        __bf16*       l = &dst[n * SBROW + part * 8];
#if USE_ASYNC_LDS
        __builtin_amdgcn_global_load_async_to_lds_b128(
            (__attribute__((address_space(1))) v4i*)g,
            (__attribute__((address_space(3))) v4i*)l,
            /*offset=*/0, /*cpol=*/0);
#else
        *(v8bf*)l = *(const v8bf*)g;
#endif
    }
}

// One N-tile step: preload next B fragment, WMMA current one.
// REUSE must be a literal so the builtin sees an integer constant expression.
#define BSTEP(NT, REUSE)                                                      \
    {                                                                         \
        v8bf nlo = blo, nhi = bhi;                                            \
        if ((NT) < 7) {                                                       \
            nlo = *(const v8bf*)&brow[((NT) + 1) * 16 * SBROW];               \
            nhi = *(const v8bf*)&brow[((NT) + 1) * 16 * SBROW + 8];           \
        }                                                                     \
        v16bf bfrag = __builtin_shufflevector(blo, bhi,                       \
            0, 1, 2, 3, 4, 5, 6, 7, 8, 9, 10, 11, 12, 13, 14, 15);            \
        acc[NT] = __builtin_amdgcn_wmma_f32_16x16x32_bf16(                    \
            false, afrag, false, bfrag, (short)0, acc[NT], (REUSE), false);   \
        blo = nlo;                                                            \
        bhi = nhi;                                                            \
    }

// ---------------------------------------------------------------------------
// Fused embed-gather + conv1d (as bf16 WMMA GEMM) + bias + pos-embed.
// Block: 256 threads = 8 waves; each wave owns a 16-token M-subtile and all
// 8 N-tiles. Ping-pong LDS B buffers with async staging of chunk ki+1 while
// computing chunk ki: one barrier per K-chunk.
// ---------------------------------------------------------------------------
__global__ __launch_bounds__(256)
void conv_embed_wmma_kernel(const int*   __restrict__ X,
                            const float* __restrict__ W_lin,
                            const float* __restrict__ b_lin,
                            const float* __restrict__ b_conv,
                            const float* __restrict__ pos,
                            const __bf16* __restrict__ Bt,
                            float* __restrict__ out) {
    __shared__ __bf16 ldsA[HALO_ROWS * SA];   // 35,904 B: embeddings (+conv halo)
    __shared__ __bf16 ldsB[2][DIM * SBROW];   // 2 x 10,240 B: ping-pong B chunks
    __shared__ int    sX[HALO_ROWS];

    const int tid   = threadIdx.x;
    const int batch = blockIdx.x >> 6;             // 64 tiles per batch
    const int tbase = (blockIdx.x & 63) * MTILE;

    // Stage token ids (halo rows map to zero-padded conv input).
    if (tid < HALO_ROWS) {
        int t = tbase + tid - 2;
        sX[tid] = ((unsigned)t < SEQ) ? X[batch * SEQ + t] : 0;
    }
    // Warm caches with the pos-table rows the epilogue will read.
    if (tid < MTILE) {
        __builtin_prefetch(&pos[(size_t)(tbase + tid) * DIM], 0, 1);
    }
    __syncthreads();

    // Kick off DMA of B chunk 0; it runs under the scattered gather below.
    stage_b_chunk(Bt, &ldsB[0][0], 0, tid);

    // Gather embeddings: emb[t][c] = W_lin[c][tok] + b_lin[c], bf16 into LDS.
    // W_lin (51 MB) is L2-resident on the 192 MB L2.
    for (int idx = tid; idx < HALO_ROWS * DIM; idx += 256) {
        int r = idx >> 7;
        int c = idx & 127;
        int t = tbase + r - 2;
        float v = 0.0f;
        if ((unsigned)t < SEQ) {
            v = W_lin[(size_t)c * VOCAB + sX[r]] + b_lin[c];
        }
        ldsA[r * SA + c] = (__bf16)v;
    }
#if USE_ASYNC_LDS
    wait_async0();
#endif
    __syncthreads();

    const int lane = tid & 31;
    const int w    = tid >> 5;    // wave id = M subtile (16 tokens each)
    const int h    = lane >> 4;   // lane half selects K sub-range
    const int m    = lane & 15;   // M row (A) / N col (B,C)

    v8f acc[8];
    #pragma unroll
    for (int nt = 0; nt < 8; ++nt) {
        acc[nt] = (v8f){0.f, 0.f, 0.f, 0.f, 0.f, 0.f, 0.f, 0.f};
    }

    for (int ki = 0; ki < NCHUNK; ++ki) {
        // Prefetch next B chunk into the other LDS buffer (async DMA).
        if (ki + 1 < NCHUNK) {
            stage_b_chunk(Bt, &ldsB[(ki + 1) & 1][0], ki + 1, tid);
        }
        const __bf16* Bcur = &ldsB[ki & 1][0];

        // A fragment (16x32 bf16): lane(h,m) holds K = 16h+{0..7}
        // and 16h+8+{0..7}; an 8-aligned K chunk never crosses a tap boundary.
        int k1 = ki * 32 + h * 16;
        int j1 = k1 >> 7, c1 = k1 & 127;
        int k2 = k1 + 8;
        int j2 = k2 >> 7, c2 = k2 & 127;
        v8bf alo = *(const v8bf*)&ldsA[(w * 16 + m + j1) * SA + c1];
        v8bf ahi = *(const v8bf*)&ldsA[(w * 16 + m + j2) * SA + c2];
        v16bf afrag = __builtin_shufflevector(alo, ahi,
            0, 1, 2, 3, 4, 5, 6, 7, 8, 9, 10, 11, 12, 13, 14, 15);

        // Software-pipelined B fragments; reuse_a hint on the 7 repeats of
        // the identical WMMA sharing the same A registers.
        const __bf16* brow = &Bcur[m * SBROW + h * 16];
        v8bf blo = *(const v8bf*)&brow[0];
        v8bf bhi = *(const v8bf*)&brow[8];
        BSTEP(0, false)
        BSTEP(1, true)
        BSTEP(2, true)
        BSTEP(3, true)
        BSTEP(4, true)
        BSTEP(5, true)
        BSTEP(6, true)
        BSTEP(7, true)

#if USE_ASYNC_LDS
        wait_async0();          // next chunk's DMA landed in LDS
#endif
        __syncthreads();        // all waves done reading Bcur / DMA visible
    }

    // Epilogue: C/D layout lane = N (mod 16), VGPR r = M row (lane-half adds 8).
    #pragma unroll
    for (int nt = 0; nt < 8; ++nt) {
        int   n    = nt * 16 + m;
        float bias = b_conv[n];
        #pragma unroll
        for (int r = 0; r < 8; ++r) {
            int    mm = r + 8 * h;
            int    t  = tbase + w * 16 + mm;
            size_t gi = ((size_t)batch * SEQ + t) * DIM + n;
            out[gi] = acc[nt][r] + bias + pos[(size_t)t * DIM + n];
        }
    }
}

// ---------------------------------------------------------------------------
extern "C" void kernel_launch(void* const* d_in, const int* in_sizes, int n_in,
                              void* d_out, int out_size, void* d_ws, size_t ws_size,
                              hipStream_t stream) {
    const int*   X      = (const int*)d_in[0];
    const float* W_lin  = (const float*)d_in[1];
    const float* b_lin  = (const float*)d_in[2];
    const float* W_conv = (const float*)d_in[3];
    const float* b_conv = (const float*)d_in[4];
    const float* pos    = (const float*)d_in[5];
    float*  out = (float*)d_out;
    __bf16* Bt  = (__bf16*)d_ws;   // 128*640 bf16 = 160 KB

    prep_b_kernel<<<(DIM * KTOT + 255) / 256, 256, 0, stream>>>(W_conv, Bt);

    const int blocks = 8 * (SEQ / MTILE);  // 8 batches x 64 tiles = 512
    conv_embed_wmma_kernel<<<blocks, 256, 0, stream>>>(
        X, W_lin, b_lin, b_conv, pos, Bt, out);
}
